// RN_16492674417248
// MI455X (gfx1250) — matrix operand order repaired and checked by reading, not compiled, not run on hardware
//
#include <hip/hip_runtime.h>

typedef __attribute__((ext_vector_type(16))) _Float16 v16h;
typedef __attribute__((ext_vector_type(8)))  float    v8f;
typedef unsigned int   u32;
typedef unsigned short u16;
typedef __attribute__((ext_vector_type(4))) u32 v4u;

__device__ __forceinline__ u16 f2h_bits(float v) {
  _Float16 h = (_Float16)v; u16 u; __builtin_memcpy(&u, &h, 2); return u;
}
__device__ __forceinline__ float h2f_bits(u16 u) {
  _Float16 h; __builtin_memcpy(&h, &u, 2); return (float)h;
}

// ---------------------------------------------------------------------------
// Direct stride-2 pad-1 3x3 conv (NCHW / OIHW)
// ---------------------------------------------------------------------------
__global__ void conv2d_s2(const float* __restrict__ in, const float* __restrict__ w,
                          const float* __restrict__ bias, float* __restrict__ out,
                          int Nb, int IC, int IH, int IW, int OC, int OH, int OW) {
  long idx = (long)blockIdx.x * blockDim.x + threadIdx.x;
  long tot = (long)Nb * OC * OH * OW;
  if (idx >= tot) return;
  int ow = (int)(idx % OW); long t = idx / OW;
  int oh = (int)(t % OH);   t /= OH;
  int oc = (int)(t % OC);   int n = (int)(t / OC);
  float acc = bias[oc];
  for (int ic = 0; ic < IC; ++ic) {
    const float* ip = in + ((size_t)n * IC + ic) * IH * IW;
    const float* wp = w + ((size_t)oc * IC + ic) * 9;
    #pragma unroll
    for (int kh = 0; kh < 3; ++kh) {
      int ih = oh * 2 - 1 + kh;
      if (ih < 0 || ih >= IH) continue;
      #pragma unroll
      for (int kw = 0; kw < 3; ++kw) {
        int iw = ow * 2 - 1 + kw;
        if (iw < 0 || iw >= IW) continue;
        acc += ip[ih * IW + iw] * wp[kh * 3 + kw];
      }
    }
  }
  out[idx] = acc;
}

// Per-channel batch stats (mean, biased var) over (N,H,W); one block per channel
__global__ void bn_stats(const float* __restrict__ x, int Nb, int C, int HW,
                         float* __restrict__ mean, float* __restrict__ var) {
  __shared__ float ss[256], sq[256];
  int c = blockIdx.x, tid = threadIdx.x;
  int per = Nb * HW;
  float s = 0.f, q = 0.f;
  for (int t = tid; t < per; t += 256) {
    int n = t / HW, p = t - n * HW;
    float v = x[((size_t)n * C + c) * HW + p];
    s += v; q += v * v;
  }
  ss[tid] = s; sq[tid] = q; __syncthreads();
  for (int st = 128; st > 0; st >>= 1) {
    if (tid < st) { ss[tid] += ss[tid + st]; sq[tid] += sq[tid + st]; }
    __syncthreads();
  }
  if (tid == 0) {
    float m = ss[0] / per;
    mean[c] = m;
    var[c] = sq[0] / per - m * m;
  }
}

__global__ void bn_relu(float* __restrict__ x, const float* __restrict__ mean,
                        const float* __restrict__ var, const float* __restrict__ g,
                        const float* __restrict__ be, int C, int HW, long tot) {
  long idx = (long)blockIdx.x * blockDim.x + threadIdx.x;
  if (idx >= tot) return;
  int c = (int)((idx / HW) % C);
  float v = (x[idx] - mean[c]) * rsqrtf(var[c] + 1e-5f) * g[c] + be[c];
  x[idx] = v > 0.f ? v : 0.f;
}

// x4 [64,256,5,5] f32 -> obj f16 [1600,256]  (obj[b, p, c] = x4[b, c, p])
__global__ void pack_obj(const float* __restrict__ x4, u16* __restrict__ obj) {
  long idx = (long)blockIdx.x * blockDim.x + threadIdx.x;
  if (idx >= 1600L * 256) return;
  int c = (int)(idx & 255);
  int row = (int)(idx >> 8);
  int b = row / 25, p = row - b * 25;
  obj[idx] = f2h_bits(x4[((size_t)b * 256 + c) * 25 + p]);
}

// Generic f32 sub-matrix -> zero-padded f16 matrix
__global__ void cvt_pad_h(const float* __restrict__ src, int ld, int col0, int R, int C,
                          u16* __restrict__ dst, int Rp, int Cp) {
  long idx = (long)blockIdx.x * blockDim.x + threadIdx.x;
  long tot = (long)Rp * Cp;
  if (idx >= tot) return;
  int r = (int)(idx / Cp), c = (int)(idx - (long)r * Cp);
  float v = (r < R && c < C) ? src[(size_t)r * ld + col0 + c] : 0.f;
  dst[idx] = f2h_bits(v);
}

// C[b,n] = gb1[n] + qst[b,:] @ gw1[n, 512:523]   (n<2000, else 0)
__global__ void qc_kernel(const float* __restrict__ qst, const float* __restrict__ gw1,
                          const float* __restrict__ gb1, float* __restrict__ Cq) {
  int idx = blockIdx.x * blockDim.x + threadIdx.x;
  if (idx >= 64 * 2048) return;
  int n = idx & 2047, b = idx >> 11;
  float v = 0.f;
  if (n < 2000) {
    v = gb1[n];
    const float* wr = gw1 + (size_t)n * 523 + 512;
    const float* q = qst + b * 11;
    #pragma unroll
    for (int k = 0; k < 11; ++k) v += q[k] * wr[k];
  }
  Cq[idx] = v;
}

// h1[(b,i,j), n] = relu(A[b*25+i, n] + B[b*25+j, n] + C[b, n])  -> f16
__global__ void combine_h1(const float* __restrict__ A, const float* __restrict__ Bm,
                           const float* __restrict__ Cq, u16* __restrict__ H) {
  long idx = (long)blockIdx.x * blockDim.x + threadIdx.x;
  if (idx >= 40000L * 2048) return;
  int n = (int)(idx & 2047);
  long row = idx >> 11;                 // [0, 40000)
  int b = (int)(row / 625);
  int p = (int)(row - (long)b * 625);
  int i = p / 25, j = p - i * 25;
  float v = A[((size_t)(b * 25 + i) << 11) + n] +
            Bm[((size_t)(b * 25 + j) << 11) + n] +
            Cq[((size_t)b << 11) + n];
  H[idx] = f2h_bits(v > 0.f ? v : 0.f);
}

// xg[b,n] = (1/625) * sum_r H[b*625+r, n]
__global__ void pair_sum(const u16* __restrict__ H, float* __restrict__ xg) {
  int idx = blockIdx.x * blockDim.x + threadIdx.x;
  if (idx >= 64 * 2048) return;
  int n = idx & 2047, b = idx >> 11;
  const u16* p = H + (size_t)b * 625 * 2048 + n;
  float s = 0.f;
  for (int r = 0; r < 625; ++r) s += h2f_bits(p[(size_t)r * 2048]);
  xg[idx] = s * (1.f / 625.f);
}

// ---------------------------------------------------------------------------
// LDS-tiled WMMA f16 GEMM: Out = act(X @ W^T + bias)
//   X : f16 [M,  Kp] row-major (Kp % 64 == 0)
//   W : f16 [Np, Kp] row-major (torch Linear layout; Np % 128 == 0)
//   Block (8 waves) computes a 128x128 tile; wave computes 32x64 via
//   2 A-frags x 4 B-frags = 8 WMMA accumulators. K staged in 64-wide LDS
//   tiles; global prefetch of tile k+1 issued before compute of tile k.
// ---------------------------------------------------------------------------
template <bool RELU, bool OUT_HALF>
__global__ __launch_bounds__(256) void gemm_f16_tiled(
    const u16* __restrict__ X, const u16* __restrict__ W,
    const float* __restrict__ bias, void* __restrict__ Out,
    int M, int N, int Np, int Kp) {
  // 72-half row stride (gcd(36,64)=4 dword swizzle -> conflict-free frag reads)
  __shared__ u16 As[128 * 72];
  __shared__ u16 Bs[128 * 72];

  const int nb = Np >> 7;
  const int bm = blockIdx.x / nb;
  const int bn = blockIdx.x - bm * nb;
  const int m0 = bm << 7;
  const int n0 = bn << 7;
  const int tid = (int)threadIdx.x;
  const int lane = tid & 31;
  const int wid = tid >> 5;
  const int wm = (wid & 3) << 5;   // wave m offset: 0,32,64,96
  const int wn = (wid >> 2) << 6;  // wave n offset: 0,64
  const int l15 = lane & 15;
  const int hsel = lane >> 4;

  // global<->LDS staging: 512 16B-chunks (128 rows x 32 halves) per K tile, 2 per thread
  const int r0 = tid >> 2,        cc0 = (tid & 3) << 3;        // row, half-offset
  const int r1 = (tid + 256) >> 2, cc1 = ((tid + 256) & 3) << 3;

  const int ktiles = Kp >> 6;
  const v4u zero4 = {0, 0, 0, 0};
  v4u pa0, pa1, pb0, pb1;
  pa0 = (m0 + r0 < M) ? *(const v4u*)(X + (size_t)(m0 + r0) * Kp + cc0) : zero4;
  pa1 = (m0 + r1 < M) ? *(const v4u*)(X + (size_t)(m0 + r1) * Kp + cc1) : zero4;
  pb0 = *(const v4u*)(W + (size_t)(n0 + r0) * Kp + cc0);
  pb1 = *(const v4u*)(W + (size_t)(n0 + r1) * Kp + cc1);

  v8f acc[8] = {};  // [s*4 + j] : s in {0,1} m-sub, j in {0..3} n-sub

  for (int kt = 0; kt < ktiles; ++kt) {
    __syncthreads();  // previous tile's readers done
    *(v4u*)(As + r0 * 72 + cc0) = pa0;
    *(v4u*)(As + r1 * 72 + cc1) = pa1;
    *(v4u*)(Bs + r0 * 72 + cc0) = pb0;
    *(v4u*)(Bs + r1 * 72 + cc1) = pb1;
    __syncthreads();

    if (kt + 1 < ktiles) {  // prefetch next K tile; latency hides behind WMMAs
      const int kb = (kt + 1) << 6;
      pa0 = (m0 + r0 < M) ? *(const v4u*)(X + (size_t)(m0 + r0) * Kp + kb + cc0) : zero4;
      pa1 = (m0 + r1 < M) ? *(const v4u*)(X + (size_t)(m0 + r1) * Kp + kb + cc1) : zero4;
      pb0 = *(const v4u*)(W + (size_t)(n0 + r0) * Kp + kb + cc0);
      pb1 = *(const v4u*)(W + (size_t)(n0 + r1) * Kp + kb + cc1);
    }

    #pragma unroll
    for (int kk = 0; kk < 64; kk += 32) {
      union { v16h v; u32 u[8]; } afr[2], bfr[4];
      // A 16x32 frag: lane<16 holds K {kk..kk+7, kk+16..kk+23}, lane>=16 the rest
      #pragma unroll
      for (int s = 0; s < 2; ++s) {
        const u32* pr = (const u32*)(As + (wm + (s << 4) + l15) * 72);
        const int ka = (kk >> 1) + (hsel << 2);
        #pragma unroll
        for (int i = 0; i < 4; ++i) { afr[s].u[i] = pr[ka + i]; afr[s].u[4 + i] = pr[ka + 8 + i]; }
      }
      // B 32x16 frag: lane<16 holds K {kk..kk+7, kk+16..kk+23}, lane>=16 the rest
      #pragma unroll
      for (int j = 0; j < 4; ++j) {
        const u32* pr = (const u32*)(Bs + (wn + (j << 4) + l15) * 72);
        const int kb2 = (kk >> 1) + (hsel << 2);
        #pragma unroll
        for (int i = 0; i < 4; ++i) { bfr[j].u[i] = pr[kb2 + i]; bfr[j].u[4 + i] = pr[kb2 + 8 + i]; }
      }
      #pragma unroll
      for (int s = 0; s < 2; ++s)
        #pragma unroll
        for (int j = 0; j < 4; ++j)
          acc[s * 4 + j] = __builtin_amdgcn_wmma_f32_16x16x32_f16(
              false, afr[s].v, false, bfr[j].v, (short)0, acc[s * 4 + j], false, false);
    }
  }

  #pragma unroll
  for (int s = 0; s < 2; ++s) {
    #pragma unroll
    for (int j = 0; j < 4; ++j) {
      const int col = n0 + wn + (j << 4) + l15;
      const float bv = (bias != nullptr && col < N) ? bias[col] : 0.f;
      #pragma unroll
      for (int r = 0; r < 8; ++r) {
        const int row = m0 + wm + (s << 4) + r + (hsel << 3);  // 32-bit C/D layout
        if (row < M) {
          float v = acc[s * 4 + j][r] + bv;
          if (RELU) v = v > 0.f ? v : 0.f;
          if (OUT_HALF) ((u16*)Out)[(size_t)row * Np + col] = f2h_bits(v);
          else          ((float*)Out)[(size_t)row * Np + col] = v;
        }
      }
    }
  }
}

// Final 100->10 layer + log_softmax; one thread per batch element
__global__ void final_fc(const float* __restrict__ xf3, const float* __restrict__ fw4,
                         const float* __restrict__ fb4, float* __restrict__ out) {
  int b = blockIdx.x * blockDim.x + threadIdx.x;
  if (b >= 64) return;
  const float* x = xf3 + (size_t)b * 128;
  float lg[10], mx = -3.0e38f;
  #pragma unroll
  for (int j = 0; j < 10; ++j) {
    float v = fb4[j];
    const float* w = fw4 + j * 100;
    for (int k = 0; k < 100; ++k) v += x[k] * w[k];
    lg[j] = v;
    mx = fmaxf(mx, v);
  }
  float s = 0.f;
  #pragma unroll
  for (int j = 0; j < 10; ++j) s += expf(lg[j] - mx);
  float lse = mx + logf(s);
  #pragma unroll
  for (int j = 0; j < 10; ++j) out[b * 10 + j] = lg[j] - lse;
}

// ---------------------------------------------------------------------------
extern "C" void kernel_launch(void* const* d_in, const int* in_sizes, int n_in,
                              void* d_out, int out_size, void* d_ws, size_t ws_size,
                              hipStream_t stream) {
  (void)in_sizes; (void)n_in; (void)out_size; (void)ws_size;

  const float* img = (const float*)d_in[0];
  const float* qst = (const float*)d_in[1];
  const float* w1 = (const float*)d_in[2];  const float* b1 = (const float*)d_in[3];
  const float* w2 = (const float*)d_in[4];  const float* b2 = (const float*)d_in[5];
  const float* w3 = (const float*)d_in[6];  const float* b3 = (const float*)d_in[7];
  const float* w4 = (const float*)d_in[8];  const float* b4 = (const float*)d_in[9];
  const float* g1 = (const float*)d_in[10]; const float* be1 = (const float*)d_in[11];
  const float* g2 = (const float*)d_in[12]; const float* be2 = (const float*)d_in[13];
  const float* g3 = (const float*)d_in[14]; const float* be3 = (const float*)d_in[15];
  const float* g4 = (const float*)d_in[16]; const float* be4 = (const float*)d_in[17];
  const float* gw1 = (const float*)d_in[18]; const float* gb1 = (const float*)d_in[19];
  const float* gw2 = (const float*)d_in[20]; const float* gb2 = (const float*)d_in[21];
  const float* gw3 = (const float*)d_in[22]; const float* gb3 = (const float*)d_in[23];
  const float* gw4 = (const float*)d_in[24]; const float* gb4 = (const float*)d_in[25];
  const float* fw1 = (const float*)d_in[26]; const float* fb1 = (const float*)d_in[27];
  const float* fw2 = (const float*)d_in[28]; const float* fb2 = (const float*)d_in[29];
  const float* fw3 = (const float*)d_in[30]; const float* fb3 = (const float*)d_in[31];
  const float* fw4 = (const float*)d_in[32]; const float* fb4 = (const float*)d_in[33];

  char* wsb = (char*)d_ws;
  size_t off = 0;
  auto alloc = [&](size_t bytes) -> char* {
    char* p = wsb + off;
    off += (bytes + 255) & ~(size_t)255;
    return p;
  };

  float* x1 = (float*)alloc(64UL * 32 * 38 * 38 * 4);
  float* x2 = (float*)alloc(64UL * 64 * 19 * 19 * 4);
  float* x3 = (float*)alloc(64UL * 128 * 10 * 10 * 4);
  float* x4 = (float*)alloc(64UL * 256 * 5 * 5 * 4);
  float* mn1 = (float*)alloc(32 * 4);  float* vr1 = (float*)alloc(32 * 4);
  float* mn2 = (float*)alloc(64 * 4);  float* vr2 = (float*)alloc(64 * 4);
  float* mn3 = (float*)alloc(128 * 4); float* vr3 = (float*)alloc(128 * 4);
  float* mn4 = (float*)alloc(256 * 4); float* vr4 = (float*)alloc(256 * 4);
  u16* obj  = (u16*)alloc(1600UL * 256 * 2);
  u16* W1a  = (u16*)alloc(2048UL * 256 * 2);
  u16* W1b  = (u16*)alloc(2048UL * 256 * 2);
  float* Amat = (float*)alloc(1600UL * 2048 * 4);
  float* Bmat = (float*)alloc(1600UL * 2048 * 4);
  float* Cq   = (float*)alloc(64UL * 2048 * 4);
  u16* gw2h = (u16*)alloc(2048UL * 2048 * 2);
  u16* gw3h = (u16*)alloc(2048UL * 2048 * 2);
  u16* gw4h = (u16*)alloc(2048UL * 2048 * 2);
  u16* hA = (u16*)alloc(40000UL * 2048 * 2);
  u16* hB = (u16*)alloc(40000UL * 2048 * 2);
  float* xg  = (float*)alloc(64UL * 2048 * 4);
  u16* xgh   = (u16*)alloc(64UL * 2048 * 2);
  u16* fw1h  = (u16*)alloc(1024UL * 2048 * 2);
  u16* fw2h  = (u16*)alloc(512UL * 1024 * 2);
  u16* fw3h  = (u16*)alloc(128UL * 512 * 2);
  float* xf1 = (float*)alloc(64UL * 1024 * 4);
  u16* xf1h  = (u16*)alloc(64UL * 1024 * 2);
  float* xf2 = (float*)alloc(64UL * 512 * 4);
  u16* xf2h  = (u16*)alloc(64UL * 512 * 2);
  float* xf3 = (float*)alloc(64UL * 128 * 4);

  auto cdiv = [](long a, long b) -> unsigned { return (unsigned)((a + b - 1) / b); };

  // ---- conv stack with training-mode BN + ReLU ----
  long t1 = 64L * 32 * 38 * 38;
  conv2d_s2<<<cdiv(t1, 256), 256, 0, stream>>>(img, w1, b1, x1, 64, 3, 75, 75, 32, 38, 38);
  bn_stats<<<32, 256, 0, stream>>>(x1, 64, 32, 1444, mn1, vr1);
  bn_relu<<<cdiv(t1, 256), 256, 0, stream>>>(x1, mn1, vr1, g1, be1, 32, 1444, t1);

  long t2 = 64L * 64 * 19 * 19;
  conv2d_s2<<<cdiv(t2, 256), 256, 0, stream>>>(x1, w2, b2, x2, 64, 32, 38, 38, 64, 19, 19);
  bn_stats<<<64, 256, 0, stream>>>(x2, 64, 64, 361, mn2, vr2);
  bn_relu<<<cdiv(t2, 256), 256, 0, stream>>>(x2, mn2, vr2, g2, be2, 64, 361, t2);

  long t3 = 64L * 128 * 10 * 10;
  conv2d_s2<<<cdiv(t3, 256), 256, 0, stream>>>(x2, w3, b3, x3, 64, 64, 19, 19, 128, 10, 10);
  bn_stats<<<128, 256, 0, stream>>>(x3, 64, 128, 100, mn3, vr3);
  bn_relu<<<cdiv(t3, 256), 256, 0, stream>>>(x3, mn3, vr3, g3, be3, 128, 100, t3);

  long t4 = 64L * 256 * 5 * 5;
  conv2d_s2<<<cdiv(t4, 256), 256, 0, stream>>>(x3, w4, b4, x4, 64, 128, 10, 10, 256, 5, 5);
  bn_stats<<<256, 256, 0, stream>>>(x4, 64, 256, 25, mn4, vr4);
  bn_relu<<<cdiv(t4, 256), 256, 0, stream>>>(x4, mn4, vr4, g4, be4, 256, 25, t4);

  pack_obj<<<cdiv(1600L * 256, 256), 256, 0, stream>>>(x4, obj);

  // ---- weight conversions to zero-padded f16 ----
  cvt_pad_h<<<cdiv(2048L * 256, 256), 256, 0, stream>>>(gw1, 523, 0, 2000, 256, W1a, 2048, 256);
  cvt_pad_h<<<cdiv(2048L * 256, 256), 256, 0, stream>>>(gw1, 523, 256, 2000, 256, W1b, 2048, 256);
  cvt_pad_h<<<cdiv(2048L * 2048, 256), 256, 0, stream>>>(gw2, 2000, 0, 2000, 2000, gw2h, 2048, 2048);
  cvt_pad_h<<<cdiv(2048L * 2048, 256), 256, 0, stream>>>(gw3, 2000, 0, 2000, 2000, gw3h, 2048, 2048);
  cvt_pad_h<<<cdiv(2048L * 2048, 256), 256, 0, stream>>>(gw4, 2000, 0, 2000, 2000, gw4h, 2048, 2048);
  cvt_pad_h<<<cdiv(1024L * 2048, 256), 256, 0, stream>>>(fw1, 2000, 0, 1000, 2000, fw1h, 1024, 2048);
  cvt_pad_h<<<cdiv(512L * 1024, 256), 256, 0, stream>>>(fw2, 1000, 0, 500, 1000, fw2h, 512, 1024);
  cvt_pad_h<<<cdiv(128L * 512, 256), 256, 0, stream>>>(fw3, 500, 0, 100, 500, fw3h, 128, 512);

  auto ggrid = [&](int M, int Np) -> unsigned {
    return (unsigned)(((M + 127) / 128) * (Np >> 7));
  };

  // ---- layer-1 factorization: A = obj@W1a^T, B = obj@W1b^T, C = qst@W1q^T + gb1 ----
  gemm_f16_tiled<false, false><<<ggrid(1600, 2048), 256, 0, stream>>>(obj, W1a, nullptr, Amat, 1600, 2048, 2048, 256);
  gemm_f16_tiled<false, false><<<ggrid(1600, 2048), 256, 0, stream>>>(obj, W1b, nullptr, Bmat, 1600, 2048, 2048, 256);
  qc_kernel<<<cdiv(64L * 2048, 256), 256, 0, stream>>>(qst, gw1, gb1, Cq);
  combine_h1<<<cdiv(40000L * 2048, 256), 256, 0, stream>>>(Amat, Bmat, Cq, hA);

  // ---- g MLP: 3x [40000x2048] @ [2048x2048] WMMA GEMMs (the 1-TFLOP core) ----
  gemm_f16_tiled<true, true><<<ggrid(40000, 2048), 256, 0, stream>>>(hA, gw2h, gb2, hB, 40000, 2000, 2048, 2048);
  gemm_f16_tiled<true, true><<<ggrid(40000, 2048), 256, 0, stream>>>(hB, gw3h, gb3, hA, 40000, 2000, 2048, 2048);
  gemm_f16_tiled<true, true><<<ggrid(40000, 2048), 256, 0, stream>>>(hA, gw4h, gb4, hB, 40000, 2000, 2048, 2048);

  // ---- pair reduction + f MLP ----
  pair_sum<<<cdiv(64L * 2048, 256), 256, 0, stream>>>(hB, xg);
  cvt_pad_h<<<cdiv(64L * 2048, 256), 256, 0, stream>>>(xg, 2048, 0, 64, 2048, xgh, 64, 2048);
  gemm_f16_tiled<true, false><<<ggrid(64, 1024), 256, 0, stream>>>(xgh, fw1h, fb1, xf1, 64, 1000, 1024, 2048);
  cvt_pad_h<<<cdiv(64L * 1024, 256), 256, 0, stream>>>(xf1, 1024, 0, 64, 1024, xf1h, 64, 1024);
  gemm_f16_tiled<true, false><<<ggrid(64, 512), 256, 0, stream>>>(xf1h, fw2h, fb2, xf2, 64, 500, 512, 1024);
  cvt_pad_h<<<cdiv(64L * 512, 256), 256, 0, stream>>>(xf2, 512, 0, 64, 512, xf2h, 64, 512);
  gemm_f16_tiled<true, false><<<ggrid(64, 128), 256, 0, stream>>>(xf2h, fw3h, fb3, xf3, 64, 100, 128, 512);

  final_fc<<<1, 64, 0, stream>>>(xf3, fw4, fb4, (float*)d_out);
}
